// SAGE_78580721648122
// MI455X (gfx1250) — matrix operand order repaired and checked by reading, not compiled, not run on hardware
//
#include <hip/hip_runtime.h>

typedef unsigned short u16;
typedef __attribute__((ext_vector_type(8)))  unsigned short v8u;
typedef __attribute__((ext_vector_type(16))) unsigned short v16u;
typedef __attribute__((ext_vector_type(16))) __bf16         v16bf;
typedef __attribute__((ext_vector_type(8)))  float          v8f;

#define F_IN  128
#define F_OUT 40
#define AST   256   // fused A-buffer row stride (bf16 elements): [x/h | mean]

static __device__ __forceinline__ u16 f2bf(float f) {
  unsigned u = __builtin_bit_cast(unsigned, f);
  u += 0x7fffu + ((u >> 16) & 1u);          // round-to-nearest-even
  return (u16)(u >> 16);
}
static __device__ __forceinline__ float bf2f(u16 h) {
  return __builtin_bit_cast(float, ((unsigned)h) << 16);
}
static __device__ __forceinline__ v16bf as_bf16(v16u u) {
  union { v16u u; v16bf b; } c; c.u = u; return c.b;
}

// ---------------------------------------------------------------- utilities
__global__ void k_zero(float* __restrict__ p, long n) {
  long i = (long)blockIdx.x * blockDim.x + threadIdx.x;
  long stride = (long)gridDim.x * blockDim.x;
  for (; i < n; i += stride) p[i] = 0.f;
}

__global__ void k_degree(const int* __restrict__ dst, float* __restrict__ deg, int E) {
  int e = blockIdx.x * blockDim.x + threadIdx.x;
  if (e < E) atomicAdd(deg + dst[e], 1.f);
}

// x (f32, N x 128) -> bf16 into A[:, 0:128]
__global__ void k_conv_x(const float* __restrict__ x, u16* __restrict__ A, long nf) {
  long i = (long)blockIdx.x * blockDim.x + threadIdx.x;
  if (i >= nf) return;
  long n = i >> 7; int f = (int)(i & 127);
  A[n * AST + f] = f2bf(x[i]);
}

// agg/deg -> bf16 into A[:, 128:256]
__global__ void k_mean_to_A(const float* __restrict__ agg, const float* __restrict__ deg,
                            u16* __restrict__ A, long nf) {
  long i = (long)blockIdx.x * blockDim.x + threadIdx.x;
  if (i >= nf) return;
  long n = i >> 7; int f = (int)(i & 127);
  float d = fmaxf(deg[n], 1.f);
  A[n * AST + 128 + f] = f2bf(agg[i] / d);
}

// edge scatter: agg[dst] += x[src]  (fp32 source), 1 wave per edge, float4/lane
__global__ void k_scatter_f32(const float* __restrict__ x, const int* __restrict__ src,
                              const int* __restrict__ dst, float* __restrict__ agg, int E) {
  int t = blockIdx.x * blockDim.x + threadIdx.x;
  int e = t >> 5; int q = (t & 31) * 4;
  if (e >= E) return;
  int s = src[e], d = dst[e];
  const float4 v = *(const float4*)(x + (size_t)s * F_IN + q);
  float* ap = agg + (size_t)d * F_IN + q;
  atomicAdd(ap + 0, v.x); atomicAdd(ap + 1, v.y);
  atomicAdd(ap + 2, v.z); atomicAdd(ap + 3, v.w);
}

// edge scatter from bf16 hidden activations stored in A[:, 0:128]
__global__ void k_scatter_bf16(const u16* __restrict__ A, const int* __restrict__ src,
                               const int* __restrict__ dst, float* __restrict__ agg, int E) {
  int t = blockIdx.x * blockDim.x + threadIdx.x;
  int e = t >> 5; int q = (t & 31) * 4;
  if (e >= E) return;
  int s = src[e], d = dst[e];
  const ushort4 v = *(const ushort4*)(A + (size_t)s * AST + q);
  float* ap = agg + (size_t)d * F_IN + q;
  atomicAdd(ap + 0, bf2f(v.x)); atomicAdd(ap + 1, bf2f(v.y));
  atomicAdd(ap + 2, bf2f(v.z)); atomicAdd(ap + 3, bf2f(v.w));
}

// Pack a (stacked) weight matrix into WMMA B-fragment order for 16x16x32 bf16.
// B layout: lane l holds column n = nt*16 + (l&15); element i -> k = kt*32 + (l>>4)*16 + i.
__global__ void k_pack_b(const float* __restrict__ Wtop, const float* __restrict__ Wbot,
                         int Ktop, int Ncols, int validN, int KT, int NT,
                         u16* __restrict__ out) {
  int t = blockIdx.x * blockDim.x + threadIdx.x;
  if (t >= KT * NT * 32) return;
  int lane = t & 31; int rest = t >> 5;
  int nt = rest % NT; int kt = rest / NT;
  int half = lane >> 4, nl = lane & 15;
  int n = nt * 16 + nl;
  u16* o = out + (size_t)t * 16;
#pragma unroll
  for (int i = 0; i < 16; i++) {
    int k = kt * 32 + half * 16 + i;
    float w = 0.f;
    if (n < validN)
      w = (k < Ktop) ? Wtop[(size_t)k * Ncols + n] : Wbot[(size_t)(k - Ktop) * Ncols + n];
    o[i] = f2bf(w);
  }
}

// ------------------------------------------- hidden GEMM: [16 x 256]·[256 x 128] per wave
// Packed B (64 KB) is staged once per workgroup into LDS; WMMAs feed from ds_load_b128.
// relu(A @ Wpacked + b) -> bf16, written in place into A[:, 0:128]
__global__ __launch_bounds__(256)
void k_gemm_hidden(const u16* __restrict__ Ain, const u16* __restrict__ pB,
                   const float* __restrict__ bias, u16* Aout, int ntiles) {
  __shared__ u16 sB[8 * 8 * 32 * 16];                 // 64 KB
  {
    const uint4* src = (const uint4*)pB;              // 4096 x 16B
    uint4* dst = (uint4*)sB;
    for (int i = threadIdx.x; i < 4096; i += 256) dst[i] = src[i];
  }
  __syncthreads();

  int wv = (blockIdx.x * blockDim.x + threadIdx.x) >> 5;
  if (wv < ntiles) {
    int lane = threadIdx.x & 31;
    int half = lane >> 4, nl = lane & 15;
    const u16* arow = Ain + ((size_t)wv * 16 + nl) * AST;
    const v16u* B = (const v16u*)sB;

    v8f acc[8] = {};
#pragma unroll
    for (int kt = 0; kt < 8; kt++) {
      // A fragment: two contiguous 16B runs per lane (k = kt*32 + half*16 + i, as in packed B)
      v8u lo = *(const v8u*)(arow + kt * 32 + half * 16);
      v8u hi = *(const v8u*)(arow + kt * 32 + half * 16 + 8);
      v16u av;
#pragma unroll
      for (int i = 0; i < 8; i++) { av[i] = lo[i]; av[i + 8] = hi[i]; }
      v16bf a = as_bf16(av);
#pragma unroll
      for (int nt = 0; nt < 8; nt++) {
        v16bf b = as_bf16(B[(kt * 8 + nt) * 32 + lane]);
        acc[nt] = __builtin_amdgcn_wmma_f32_16x16x32_bf16(
            false, a, false, b, (short)0, acc[nt], false, false);
      }
    }
    // epilogue: C layout -> row = vgpr + 8*half, col = nt*16 + nl
#pragma unroll
    for (int nt = 0; nt < 8; nt++) {
      int n = nt * 16 + nl;
      float bv = bias[n];
#pragma unroll
      for (int v = 0; v < 8; v++) {
        size_t r = (size_t)wv * 16 + v + half * 8;
        Aout[r * AST + n] = f2bf(fmaxf(acc[nt][v] + bv, 0.f));
      }
    }
  }
}

// ------------------------------ final GEMM [16 x 128]·[128 x 48] + fused log_softmax(40)
__global__ __launch_bounds__(256)
void k_gemm_final(const u16* __restrict__ Ain, const u16* __restrict__ pB,
                  const float* __restrict__ bias, float* __restrict__ out, int ntiles) {
  __shared__ u16 sB[4 * 3 * 32 * 16];                 // 12 KB
  {
    const uint4* src = (const uint4*)pB;              // 768 x 16B
    uint4* dst = (uint4*)sB;
    for (int i = threadIdx.x; i < 768; i += 256) dst[i] = src[i];
  }
  __syncthreads();

  int wv = (blockIdx.x * blockDim.x + threadIdx.x) >> 5;
  if (wv < ntiles) {
    int lane = threadIdx.x & 31;
    int half = lane >> 4, nl = lane & 15;
    const u16* arow = Ain + ((size_t)wv * 16 + nl) * AST;
    const v16u* B = (const v16u*)sB;

    v8f acc[3] = {};
#pragma unroll
    for (int kt = 0; kt < 4; kt++) {
      v8u lo = *(const v8u*)(arow + kt * 32 + half * 16);
      v8u hi = *(const v8u*)(arow + kt * 32 + half * 16 + 8);
      v16u av;
#pragma unroll
      for (int i = 0; i < 8; i++) { av[i] = lo[i]; av[i + 8] = hi[i]; }
      v16bf a = as_bf16(av);
#pragma unroll
      for (int nt = 0; nt < 3; nt++) {
        v16bf b = as_bf16(B[(kt * 3 + nt) * 32 + lane]);
        acc[nt] = __builtin_amdgcn_wmma_f32_16x16x32_bf16(
            false, a, false, b, (short)0, acc[nt], false, false);
      }
    }
    // fused log_softmax over 40 classes; row of C lives across 16 lanes (fixed half)
#pragma unroll
    for (int v = 0; v < 8; v++) {
      float lg[3];
#pragma unroll
      for (int nt = 0; nt < 3; nt++) {
        int n = nt * 16 + nl;
        lg[nt] = (n < F_OUT) ? (acc[nt][v] + bias[n]) : -__builtin_inff();
      }
      float m = fmaxf(fmaxf(lg[0], lg[1]), lg[2]);
#pragma unroll
      for (int off = 8; off > 0; off >>= 1) m = fmaxf(m, __shfl_xor(m, off, 16));
      float s = expf(lg[0] - m) + expf(lg[1] - m) + expf(lg[2] - m);
#pragma unroll
      for (int off = 8; off > 0; off >>= 1) s += __shfl_xor(s, off, 16);
      float lse = m + logf(s);
      size_t r = (size_t)wv * 16 + v + half * 8;
#pragma unroll
      for (int nt = 0; nt < 3; nt++) {
        int n = nt * 16 + nl;
        if (n < F_OUT) out[r * F_OUT + n] = lg[nt] - lse;
      }
    }
  }
}

// ---------------------------------------------------------------- launcher
extern "C" void kernel_launch(void* const* d_in, const int* in_sizes, int n_in,
                              void* d_out, int out_size, void* d_ws, size_t ws_size,
                              hipStream_t stream) {
  (void)n_in; (void)out_size; (void)ws_size;
  const float* x   = (const float*)d_in[0];
  const int*   eds = (const int*)d_in[1];
  const int*   esr = (const int*)d_in[2];
  const float* W1s = (const float*)d_in[3];
  const float* W1n = (const float*)d_in[4];
  const float* b1  = (const float*)d_in[5];
  const float* W2s = (const float*)d_in[6];
  const float* W2n = (const float*)d_in[7];
  const float* b2  = (const float*)d_in[8];
  const float* Wm  = (const float*)d_in[9];
  const float* bm  = (const float*)d_in[10];
  float* out = (float*)d_out;

  const int N = in_sizes[0] / F_IN;
  const int E = in_sizes[1];
  const int ntiles = (N + 15) / 16;   // N=50000 -> 3125 exact

  char* ws = (char*)d_ws;
  size_t off = 0;
  auto alloc = [&](size_t bytes) -> char* {
    char* p = ws + off; off = (off + bytes + 255) & ~(size_t)255; return p;
  };
  float* deg = (float*)alloc((size_t)N * 4);
  float* agg = (float*)alloc((size_t)N * F_IN * 4);
  u16*   A   = (u16*)alloc((size_t)N * AST * 2);
  u16*   pB1 = (u16*)alloc((size_t)8 * 8 * 32 * 16 * 2);
  u16*   pB2 = (u16*)alloc((size_t)8 * 8 * 32 * 16 * 2);
  u16*   pBm = (u16*)alloc((size_t)4 * 3 * 32 * 16 * 2);

  const dim3 blk(256);
  const long nf   = (long)N * F_IN;
  const int gNF   = (int)((nf + 255) / 256);
  const int gE    = (E + 255) / 256;
  const int gE32  = (int)(((long)E * 32 + 255) / 256);
  const int gGemm = (ntiles * 32 + 255) / 256;

  // degrees + input conversion + weight packing
  k_zero<<<1024, blk, 0, stream>>>(deg, N);
  k_zero<<<2048, blk, 0, stream>>>(agg, nf);
  k_degree<<<gE, blk, 0, stream>>>(eds, deg, E);
  k_conv_x<<<gNF, blk, 0, stream>>>(x, A, nf);
  k_pack_b<<<(8 * 8 * 32 + 255) / 256, blk, 0, stream>>>(W1s, W1n, 128, 128, 128, 8, 8, pB1);
  k_pack_b<<<(8 * 8 * 32 + 255) / 256, blk, 0, stream>>>(W2s, W2n, 128, 128, 128, 8, 8, pB2);
  k_pack_b<<<(4 * 3 * 32 + 255) / 256, blk, 0, stream>>>(Wm, nullptr, 128, F_OUT, F_OUT, 4, 3, pBm);

  // layer 1: scatter -> mean -> WMMA GEMM (h1 bf16 written in place over x part of A)
  k_scatter_f32<<<gE32, blk, 0, stream>>>(x, esr, eds, agg, E);
  k_mean_to_A<<<gNF, blk, 0, stream>>>(agg, deg, A, nf);
  k_gemm_hidden<<<gGemm, blk, 0, stream>>>(A, pB1, b1, A, ntiles);

  // layer 2
  k_zero<<<2048, blk, 0, stream>>>(agg, nf);
  k_scatter_bf16<<<gE32, blk, 0, stream>>>(A, esr, eds, agg, E);
  k_mean_to_A<<<gNF, blk, 0, stream>>>(agg, deg, A, nf);
  k_gemm_hidden<<<gGemm, blk, 0, stream>>>(A, pB2, b2, A, ntiles);

  // head + fused log_softmax
  k_gemm_final<<<gGemm, blk, 0, stream>>>(A, pBm, bm, out, ntiles);
}
